// PtrAttention_5763846111529
// MI455X (gfx1250) — matrix operand order repaired and checked by reading, not compiled, not run on hardware
//
#include <hip/hip_runtime.h>
#include <hip/hip_bf16.h>

typedef __attribute__((ext_vector_type(16))) _Float16 v16h;
typedef __attribute__((ext_vector_type(8)))  float    v8f;

#define BB 8
#define TT 128
#define SS 256
#define LL 256
#define DD 512
#define TPB 4   // t-rows per block in the fused score kernel (enc_t reuse factor)

__device__ __forceinline__ float fast_tanh(float x) {
#if __has_builtin(__builtin_amdgcn_tanhf)
    return __builtin_amdgcn_tanhf(x);          // gfx1250 v_tanh_f32 (TRANS op)
#else
    x = fminf(fmaxf(x, -15.f), 15.f);
    float e = __expf(2.f * x);
    return (e - 1.f) / (e + 1.f);
#endif
}

// ---------------------------------------------------------------------------
// Y[m, n] = sum_k X[m, k] * W[n, k]   (i.e. X @ W^T), f32 in, f32 out,
// via v_wmma_f32_16x16x32_f16 with on-the-fly f32->f16 conversion.
// One wave per 16x16 output tile; K stepped by 32, unrolled x2.
// ---------------------------------------------------------------------------
__global__ void proj_wmma(const float* __restrict__ X, const float* __restrict__ W,
                          float* __restrict__ Y) {
    const int lane  = threadIdx.x & 31;
    const int half_ = lane >> 4;        // which 16-lane half
    const int mn    = lane & 15;        // row (A) / col (B) index within tile
    const int tileN = blockIdx.x & 15;  // LL/16 == 16 column tiles
    const int tileM = blockIdx.x >> 4;
    const int row0  = tileM * 16;
    const int col0  = tileN * 16;

    v8f acc = {};
    const float* arow = X + (size_t)(row0 + mn) * DD;
    const float* brow = W + (size_t)(col0 + mn) * DD;

#pragma unroll 2
    for (int k0 = 0; k0 < DD; k0 += 32) {
        // A 16x32 f16 fragment: lane half 0 -> K = {0..7, 16..23}, half 1 -> {8..15, 24..31}
        const float4* pa0 = (const float4*)(arow + k0 + half_ * 8);
        const float4* pa1 = (const float4*)(arow + k0 + 16 + half_ * 8);
        float4 a0 = pa0[0], a1 = pa0[1], a2 = pa1[0], a3 = pa1[1];
        // B 32x16 f16 fragment: lane half 0 -> K = {0..7, 16..23}, half 1 -> {8..15, 24..31}
        const float4* pb0 = (const float4*)(brow + k0 + half_ * 8);
        const float4* pb1 = (const float4*)(brow + k0 + 16 + half_ * 8);
        float4 b0 = pb0[0], b1 = pb0[1], b2 = pb1[0], b3 = pb1[1];

        float av[16] = { a0.x, a0.y, a0.z, a0.w, a1.x, a1.y, a1.z, a1.w,
                         a2.x, a2.y, a2.z, a2.w, a3.x, a3.y, a3.z, a3.w };
        float bv[16] = { b0.x, b0.y, b0.z, b0.w, b1.x, b1.y, b1.z, b1.w,
                         b2.x, b2.y, b2.z, b2.w, b3.x, b3.y, b3.z, b3.w };
        v16h af, bf;
#pragma unroll
        for (int e = 0; e < 16; ++e) { af[e] = (_Float16)av[e]; bf[e] = (_Float16)bv[e]; }

        acc = __builtin_amdgcn_wmma_f32_16x16x32_f16(
            /*neg_a=*/false, af, /*neg_b=*/false, bf,
            /*c_mod=*/(short)0, acc, /*reuse_a=*/false, /*reuse_b=*/false);
    }

    // C/D layout: lanes 0-15: (M=r, N=lane); lanes 16-31: (M=8+r, N=lane-16)
#pragma unroll
    for (int r = 0; r < 8; ++r) {
        const int row = row0 + r + half_ * 8;
        Y[(size_t)row * LL + col0 + mn] = acc[r];
    }
}

// ---------------------------------------------------------------------------
// Block handles TPB consecutive t-rows of one batch b:
//   scores[tt][s] = sum_l vt[l]*tanh(dec_t[b,t0+tt,l] + enc_t[b,s,l])
// then masked log-softmax over s for each row. Each enc_t float4 load feeds
// TPB accumulators (4x L2-traffic reduction). dec rows + vt live in VGPRs.
// 8 waves; wave w owns s = 8*it + w; lane owns l = {4*lane+i, 128+4*lane+i}.
// ---------------------------------------------------------------------------
__global__ void score_softmax(const float* __restrict__ dec_t,
                              const float* __restrict__ enc_t,
                              const float* __restrict__ mask,
                              const float* __restrict__ vt,
                              float* __restrict__ out) {
    const int bt0  = blockIdx.x * TPB;   // first b*T+t row (TPB divides T)
    const int b    = bt0 / TT;
    const int tid  = threadIdx.x;
    const int wave = tid >> 5;
    const int lane = tid & 31;

    __shared__ float scores[TPB][SS];
    __shared__ float redA[8];
    __shared__ float redB[8];

    const float4* vr = (const float4*)vt;
    const float4  v0 = vr[lane];
    const float4  v1 = vr[32 + lane];

    float4 d0[TPB], d1[TPB];
#pragma unroll
    for (int tt = 0; tt < TPB; ++tt) {
        const float4* dr = (const float4*)(dec_t + (size_t)(bt0 + tt) * LL);
        d0[tt] = dr[lane];
        d1[tt] = dr[32 + lane];
    }

    const float* encb = enc_t + (size_t)b * SS * LL;

    for (int it = 0; it < SS / 8; ++it) {
        const int s = it * 8 + wave;
        const float4* er = (const float4*)(encb + (size_t)s * LL);
        const float4 e0 = er[lane];
        const float4 e1 = er[32 + lane];

        float acc[TPB];
#pragma unroll
        for (int tt = 0; tt < TPB; ++tt) {
            float a;
            a  = v0.x * fast_tanh(d0[tt].x + e0.x);
            a += v0.y * fast_tanh(d0[tt].y + e0.y);
            a += v0.z * fast_tanh(d0[tt].z + e0.z);
            a += v0.w * fast_tanh(d0[tt].w + e0.w);
            a += v1.x * fast_tanh(d1[tt].x + e1.x);
            a += v1.y * fast_tanh(d1[tt].y + e1.y);
            a += v1.z * fast_tanh(d1[tt].z + e1.z);
            a += v1.w * fast_tanh(d1[tt].w + e1.w);
            acc[tt] = a;
        }
#pragma unroll
        for (int tt = 0; tt < TPB; ++tt) {
#pragma unroll
            for (int o = 16; o; o >>= 1) acc[tt] += __shfl_xor(acc[tt], o, 32);
            if (lane == 0) scores[tt][s] = acc[tt];
        }
    }
    __syncthreads();

    // masked log-softmax over S for each of the TPB rows (one thread per s)
    const int s = tid;
    for (int tt = 0; tt < TPB; ++tt) {
        float logit = scores[tt][s] + __logf(mask[(size_t)(bt0 + tt) * SS + s] + 1e-8f);

        float mx = logit;
#pragma unroll
        for (int o = 16; o; o >>= 1) mx = fmaxf(mx, __shfl_xor(mx, o, 32));
        if (lane == 0) redA[wave] = mx;
        __syncthreads();
        if (tid == 0) {
            float g = redA[0];
#pragma unroll
            for (int w = 1; w < 8; ++w) g = fmaxf(g, redA[w]);
            redA[0] = g;
        }
        __syncthreads();
        const float gmax = redA[0];

        float sum = __expf(logit - gmax);
#pragma unroll
        for (int o = 16; o; o >>= 1) sum += __shfl_xor(sum, o, 32);
        if (lane == 0) redB[wave] = sum;
        __syncthreads();
        if (tid == 0) {
            float g = 0.f;
#pragma unroll
            for (int w = 0; w < 8; ++w) g += redB[w];
            redB[0] = g;
        }
        __syncthreads();
        const float gsum = redB[0];

        out[(size_t)(bt0 + tt) * SS + s] = logit - gmax - __logf(gsum);
        __syncthreads();   // protect redA/redB before next row reuses them
    }
}

extern "C" void kernel_launch(void* const* d_in, const int* in_sizes, int n_in,
                              void* d_out, int out_size, void* d_ws, size_t ws_size,
                              hipStream_t stream) {
    const float* dec  = (const float*)d_in[0];  // (8,128,512)
    const float* enc  = (const float*)d_in[1];  // (8,256,512)
    const float* mask = (const float*)d_in[2];  // (8,128,256)
    const float* W1   = (const float*)d_in[3];  // (256,512)
    const float* W2   = (const float*)d_in[4];  // (256,512)
    const float* vt   = (const float*)d_in[5];  // (256,)
    float* out = (float*)d_out;                 // (8,128,256)

    float* dec_t = (float*)d_ws;                        // B*T*L floats (1 MB)
    float* enc_t = dec_t + (size_t)BB * TT * LL;        // B*S*L floats (2 MB)

    // dec_t = dec @ W1^T : 64 row tiles x 16 col tiles
    proj_wmma<<<(BB * TT / 16) * (LL / 16), 32, 0, stream>>>(dec, W1, dec_t);
    // enc_t = enc @ W2^T : 128 row tiles x 16 col tiles
    proj_wmma<<<(BB * SS / 16) * (LL / 16), 32, 0, stream>>>(enc, W2, enc_t);
    // fused tanh-dot + masked log-softmax, TPB t-rows per block
    score_softmax<<<BB * TT / TPB, 256, 0, stream>>>(dec_t, enc_t, mask, vt, out);
}
